// GATv2TrajectoryPredictor_22625887715581
// MI455X (gfx1250) — matrix operand order repaired and checked by reading, not compiled, not run on hardware
//
#include <hip/hip_runtime.h>
#include <hip/hip_bf16.h>

typedef __attribute__((ext_vector_type(16))) _Float16 v16h;
typedef __attribute__((ext_vector_type(8)))  _Float16 v8h;
typedef __attribute__((ext_vector_type(8)))  float    v8f;

// ---------------------------------------------------------------------------
// WMMA GEMM on preconverted f16 operands:
//   C[M,Nc] = act( Ah[M,Kp] @ Wt[Nc,Kp]^T + bias ),  Kp % 32 == 0 (zero-padded)
// Each wave32 computes one 16x16 tile with v_wmma_f32_16x16x32_f16.
// A fragment (16-bit A 16x32): lane holds row lane&15; K-base kbA = (lane>=16)*8;
//   halves = [k0+kbA .. +7] ++ [k0+16+kbA .. +7]  -> two aligned b128 loads.
// B fragment (16-bit B 32x16): lane holds col lane&15; kbB = (lane>=16)*8;
//   halves = Wt[col, k0+kbB .. +7] ++ Wt[col, k0+16+kbB .. +7] -> two aligned b128 loads.
// C layout: vgpr v -> row 16*tm + v + 8*(lane>=16), col 16*tn + (lane&15).
// OOB rows/cols are index-clamped only (their C rows/cols are never stored).
// act: 0 = none, 1 = relu
// ---------------------------------------------------------------------------
__global__ __launch_bounds__(256) void k_gemm(const _Float16* __restrict__ Ah,
                                              const _Float16* __restrict__ Wt,
                                              const float* __restrict__ bias,
                                              float* __restrict__ C,
                                              int M, int Kp, int Nc, int act,
                                              int tiles)
{
    int wid  = (int)((blockIdx.x * blockDim.x + threadIdx.x) >> 5);
    if (wid >= tiles) return;                 // wave-uniform exit (EXEC stays all-1)
    int lane = (int)(threadIdx.x & 31);
    int tilesN = (Nc + 15) >> 4;
    int tm = wid / tilesN;
    int tn = wid - tm * tilesN;

    int row  = tm * 16 + (lane & 15);
    int colB = tn * 16 + (lane & 15);
    int rowC = (row  < M)  ? row  : 0;        // clamp: OOB row never stored
    int colC = (colB < Nc) ? colB : 0;        // clamp: OOB col never stored
    int kbA = (lane >> 4) * 8;                // A K-base: 0 or 8
    int kbB = (lane >> 4) * 8;                // B K-base: 0 or 8

    const _Float16* Arow = Ah + (size_t)rowC * Kp + kbA;
    const _Float16* Brow = Wt + (size_t)colC * Kp + kbB;

    v8f acc = {};
    for (int k0 = 0; k0 < Kp; k0 += 32) {
        v8h a0 = *(const v8h*)(Arow + k0);        // 16B aligned
        v8h a1 = *(const v8h*)(Arow + k0 + 16);
        v8h b0 = *(const v8h*)(Brow + k0);
        v8h b1 = *(const v8h*)(Brow + k0 + 16);
        v16h af = __builtin_shufflevector(a0, a1, 0,1,2,3,4,5,6,7,
                                                  8,9,10,11,12,13,14,15);
        v16h bf = __builtin_shufflevector(b0, b1, 0,1,2,3,4,5,6,7,
                                                  8,9,10,11,12,13,14,15);
        acc = __builtin_amdgcn_wmma_f32_16x16x32_f16(
                  false, af, false, bf, (short)0, acc, false, false);
    }

    int crow = tm * 16 + (lane >> 4) * 8;
    int ccol = tn * 16 + (lane & 15);
    if (ccol < Nc) {
        float bv = bias ? bias[ccol] : 0.0f;
#pragma unroll
        for (int v = 0; v < 8; ++v) {
            int r = crow + v;
            if (r < M) {
                float val = acc[v] + bv;
                if (act == 1) val = fmaxf(val, 0.0f);
                C[(size_t)r * Nc + ccol] = val;
            }
        }
    }
}

// A[M,K] f32 -> Ah[M,Kp] f16, zero-padded K..Kp
__global__ void k_cvt_pad(const float* __restrict__ X, _Float16* __restrict__ Xh,
                          int K, int Kp, long tot) {
    long i = blockIdx.x * (long)blockDim.x + threadIdx.x;
    if (i >= tot) return;
    long r = i / Kp;
    int  k = (int)(i - r * Kp);
    Xh[i] = (k < K) ? (_Float16)X[r * K + k] : (_Float16)0.0f;
}

// W[K,Nc] f32 -> Wt[Nc,Kp] f16 (transposed), zero-padded K..Kp
__global__ void k_cvt_tr_pad(const float* __restrict__ W, _Float16* __restrict__ Wt,
                             int K, int Nc, int Kp, long tot) {
    long i = blockIdx.x * (long)blockDim.x + threadIdx.x;
    if (i >= tot) return;
    long c = i / Kp;
    int  k = (int)(i - c * Kp);
    Wt[i] = (k < K) ? (_Float16)W[(size_t)k * Nc + c] : (_Float16)0.0f;
}

// ---------------------------------------------------------------------------
// Small elementwise / edge kernels (L2-resident scatter/gather work)
// ---------------------------------------------------------------------------
__global__ void k_fill_u32(unsigned* __restrict__ p, unsigned v, long n) {
    long i = blockIdx.x * (long)blockDim.x + threadIdx.x;
    if (i < n) p[i] = v;
}

__global__ void k_bias_init(float* __restrict__ out, const float* __restrict__ b,
                            int dim, long n) {
    long i = blockIdx.x * (long)blockDim.x + threadIdx.x;
    if (i < n) out[i] = b[(int)(i % dim)];
}

__global__ void k_elu(float* __restrict__ p, long n) {
    long i = blockIdx.x * (long)blockDim.x + threadIdx.x;
    if (i < n) {
        float v = p[i];
        p[i] = v > 0.0f ? v : (__expf(v) - 1.0f);
    }
}

// Pass A: g[e,h] = sum_c a[h,c]*leaky_relu(xl[src]+xr[dst]); atomic-max into gmax
__global__ void k_edge_g(const int* __restrict__ ei, int E, int Nn, int Hh,
                         const float* __restrict__ xl, const float* __restrict__ xr,
                         const float* __restrict__ av,
                         float* __restrict__ g, unsigned* __restrict__ gmaxU)
{
    long i = blockIdx.x * (long)blockDim.x + threadIdx.x;
    long tot = (long)(E + Nn) * Hh;
    if (i >= tot) return;
    int h = (int)(i % Hh);
    int e = (int)(i / Hh);
    int s, d;
    if (e < E) { s = ei[e]; d = ei[E + e]; } else { s = d = e - E; }
    int dim = Hh * 64;
    const float* pl = xl + (size_t)s * dim + h * 64;
    const float* pr = xr + (size_t)d * dim + h * 64;
    const float* pa = av + h * 64;
    float acc = 0.0f;
#pragma unroll 4
    for (int c = 0; c < 64; ++c) {
        float v = pl[c] + pr[c];
        v = (v > 0.0f) ? v : 0.2f * v;
        acc += pa[c] * v;
    }
    g[(size_t)e * Hh + h] = acc;
    unsigned bits = __float_as_uint(acc);
    unsigned enc  = (bits & 0x80000000u) ? ~bits : (bits | 0x80000000u);
    atomicMax(&gmaxU[(size_t)d * Hh + h], enc);
}

// Pass B: ex = exp(g - gmax[dst]); atomicAdd into denom; g <- ex
__global__ void k_edge_ex(const int* __restrict__ ei, int E, int Nn, int Hh,
                          float* __restrict__ g, const unsigned* __restrict__ gmaxU,
                          float* __restrict__ gsum)
{
    long i = blockIdx.x * (long)blockDim.x + threadIdx.x;
    long tot = (long)(E + Nn) * Hh;
    if (i >= tot) return;
    int h = (int)(i % Hh);
    int e = (int)(i / Hh);
    int d = (e < E) ? ei[E + e] : (e - E);
    unsigned u = gmaxU[(size_t)d * Hh + h];
    unsigned bits = (u & 0x80000000u) ? (u & 0x7FFFFFFFu) : ~u;
    float gm = __uint_as_float(bits);
    float ex = __expf(g[(size_t)e * Hh + h] - gm);
    g[(size_t)e * Hh + h] = ex;
    atomicAdd(&gsum[(size_t)d * Hh + h], ex);
}

// Pass C: out[dst] += (ex/denom) * xl[src]; 4 channels per thread (float4 load)
__global__ void k_edge_scatter(const int* __restrict__ ei, int E, int Nn, int Hh,
                               const float* __restrict__ xl,
                               const float* __restrict__ ex,
                               const float* __restrict__ gsum,
                               float* __restrict__ out)
{
    long i = blockIdx.x * (long)blockDim.x + threadIdx.x;
    long tot = (long)(E + Nn) * Hh * 16;
    if (i >= tot) return;
    int c4 = (int)(i % 16) * 4;
    long eh = i / 16;
    int h = (int)(eh % Hh);
    int e = (int)(eh / Hh);
    int s, d;
    if (e < E) { s = ei[e]; d = ei[E + e]; } else { s = d = e - E; }
    int dim = Hh * 64;
    float alpha = ex[(size_t)e * Hh + h] / gsum[(size_t)d * Hh + h];
    const float4 v = *(const float4*)(xl + (size_t)s * dim + h * 64 + c4);
    float* po = out + (size_t)d * dim + h * 64 + c4;
    atomicAdd(po + 0, alpha * v.x);
    atomicAdd(po + 1, alpha * v.y);
    atomicAdd(po + 2, alpha * v.z);
    atomicAdd(po + 3, alpha * v.w);
}

// Column sums of m[L,dim] into mean[dim] (mean pre-zeroed; caller scales by 1/L)
__global__ void k_colsum(const float* __restrict__ m, int L, int dim,
                         float* __restrict__ mean) {
    int tid = (int)(blockIdx.x * blockDim.x + threadIdx.x);
    int total = (int)(gridDim.x * blockDim.x);
    int c = tid % dim;
    int stride = total / dim;
    int r0 = tid / dim;
    float s = 0.0f;
    for (int r = r0; r < L; r += stride) s += m[(size_t)r * dim + c];
    atomicAdd(&mean[c], s);
}

// centerline [F,P,2] -> mean over P -> out[F,2]
__global__ void k_clmean(const float* __restrict__ cl, int F, int P,
                         float* __restrict__ out) {
    int i = (int)(blockIdx.x * blockDim.x + threadIdx.x);
    if (i >= F * 2) return;
    int f = i >> 1, d = i & 1;
    float s = 0.0f;
    for (int p = 0; p < P; ++p) s += cl[((size_t)f * P + p) * 2 + d];
    out[i] = s / (float)P;
}

// comb[F,192] = [ h3[focal], map_mean, cl_feat ]
__global__ void k_comb(const float* __restrict__ h3, const int* __restrict__ focal,
                       const float* __restrict__ mean, float invL,
                       const float* __restrict__ clf, float* __restrict__ comb,
                       int F) {
    int i = (int)(blockIdx.x * blockDim.x + threadIdx.x);
    if (i >= F * 192) return;
    int f = i / 192, j = i - f * 192;
    float v;
    if (j < 64)       v = h3[(size_t)focal[f] * 64 + j];
    else if (j < 128) v = mean[j - 64] * invL;
    else              v = clf[(size_t)f * 64 + (j - 128)];
    comb[i] = v;
}

// ---------------------------------------------------------------------------
// Host-side helpers
// ---------------------------------------------------------------------------
static inline unsigned gsz(long n, int b) { return (unsigned)((n + b - 1) / b); }
static inline int kpad(int K) { return (K + 31) & ~31; }

// Convert+pad A and W, then run WMMA GEMM. XH/WT are f16 scratch.
static void launch_gemm(hipStream_t s, const float* A, const float* W,
                        const float* bias, float* C, int M, int K, int Nc, int act,
                        _Float16* XH, _Float16* WT, int cvtA) {
    int Kp = kpad(K);
    if (cvtA) {
        long ta = (long)M * Kp;
        k_cvt_pad<<<gsz(ta, 256), 256, 0, s>>>(A, XH, K, Kp, ta);
    }
    long tw = (long)Nc * Kp;
    k_cvt_tr_pad<<<gsz(tw, 256), 256, 0, s>>>(W, WT, K, Nc, Kp, tw);
    int tilesM = (M + 15) / 16, tilesN = (Nc + 15) / 16;
    long tiles = (long)tilesM * tilesN;
    unsigned grid = (unsigned)((tiles * 32 + 255) / 256);
    k_gemm<<<grid, 256, 0, s>>>(XH, WT, bias, C, M, Kp, Nc, act, (int)tiles);
}

// Full GATv2 layer (self-loops appended as edges E..E+Nn-1)
static void run_gat(hipStream_t s, const float* X, int din, int Hh,
                    const float* Wl, const float* Wr, const float* av,
                    const float* bias, const int* ei, int E, int Nn,
                    float* xl, float* xr, float* g, unsigned* gmaxU, float* gsum,
                    float* out, int apply_elu, _Float16* XH, _Float16* WT) {
    int dim = Hh * 64;
    launch_gemm(s, X, Wl, nullptr, xl, Nn, din, dim, 0, XH, WT, 1); // converts X
    launch_gemm(s, X, Wr, nullptr, xr, Nn, din, dim, 0, XH, WT, 0); // reuses XH
    long nh = (long)Nn * Hh;
    k_fill_u32<<<gsz(nh, 256), 256, 0, s>>>(gmaxU, 0u, nh);
    k_fill_u32<<<gsz(nh, 256), 256, 0, s>>>((unsigned*)gsum, 0u, nh);
    long teh = (long)(E + Nn) * Hh;
    k_edge_g  <<<gsz(teh, 256), 256, 0, s>>>(ei, E, Nn, Hh, xl, xr, av, g, gmaxU);
    k_edge_ex <<<gsz(teh, 256), 256, 0, s>>>(ei, E, Nn, Hh, g, gmaxU, gsum);
    long nd = (long)Nn * dim;
    k_bias_init<<<gsz(nd, 256), 256, 0, s>>>(out, bias, dim, nd);
    long tsc = teh * 16;
    k_edge_scatter<<<gsz(tsc, 256), 256, 0, s>>>(ei, E, Nn, Hh, xl, g, gsum, out);
    if (apply_elu)
        k_elu<<<gsz(nd, 256), 256, 0, s>>>(out, nd);
}

extern "C" void kernel_launch(void* const* d_in, const int* in_sizes, int n_in,
                              void* d_out, int out_size, void* d_ws, size_t ws_size,
                              hipStream_t stream) {
    (void)n_in; (void)out_size; (void)ws_size;
    // ---- inputs (setup_inputs order) ----
    const float* x       = (const float*)d_in[0];   // [N,30,2] -> [N,60]
    const int*   ei      = (const int*)  d_in[1];   // [2,E]
    const int*   focal   = (const int*)  d_in[2];   // [F]
    const float* lane_x  = (const float*)d_in[3];   // [L,2]
    const int*   lei     = (const int*)  d_in[4];   // [2,EL]
    const float* cl      = (const float*)d_in[5];   // [F,P,2]
    const float* embW = (const float*)d_in[6],  *embB = (const float*)d_in[7];
    const float* Wl1  = (const float*)d_in[8],  *Wr1 = (const float*)d_in[9];
    const float* a1   = (const float*)d_in[10], *b1  = (const float*)d_in[11];
    const float* Wl2  = (const float*)d_in[12], *Wr2 = (const float*)d_in[13];
    const float* a2   = (const float*)d_in[14], *b2  = (const float*)d_in[15];
    const float* Wl3  = (const float*)d_in[16], *Wr3 = (const float*)d_in[17];
    const float* a3   = (const float*)d_in[18], *b3  = (const float*)d_in[19];
    const float* mWe  = (const float*)d_in[20], *mbe = (const float*)d_in[21];
    const float* Wlm  = (const float*)d_in[22], *Wrm = (const float*)d_in[23];
    const float* am   = (const float*)d_in[24], *bm  = (const float*)d_in[25];
    const float* clW  = (const float*)d_in[26], *clb = (const float*)d_in[27];
    const float* f1W  = (const float*)d_in[28], *f1b = (const float*)d_in[29];
    const float* f2W  = (const float*)d_in[30], *f2b = (const float*)d_in[31];
    const float* foW  = (const float*)d_in[32], *fob = (const float*)d_in[33];
    float* out = (float*)d_out;

    const int N  = in_sizes[0] / 60;
    const int E  = in_sizes[1] / 2;
    const int F  = in_sizes[2];
    const int L  = in_sizes[3] / 2;
    const int EL = in_sizes[4] / 2;
    const int P  = in_sizes[5] / (F * 2);

    // ---- workspace layout (floats), reuse-friendly ----
    float* ws = (float*)d_ws;
    size_t o = 0;
    auto alloc = [&](size_t n) { float* p = ws + o; o += n; return p; };
    float*    A     = alloc((size_t)N * 256);            // xl / lane mx / comb(lo)
    float*    Bb    = alloc((size_t)N * 256);            // xr / lane xl / comb(hi)
    float*    F0    = alloc((size_t)N * 256);            // feat ping / lane xr / clf / z2
    float*    F1    = alloc((size_t)N * 256);            // feat pong / lane mf / z1
    float*    H3    = alloc((size_t)N * 64);             // layer-3 output
    float*    G     = alloc(((size_t)E + N) * 4);        // per-edge g / ex (max H=4)
    unsigned* GMAXU = (unsigned*)alloc((size_t)N * 4);
    float*    GSUM  = alloc((size_t)N * 4);
    float*    MEAN  = alloc(64);
    float*    CLM   = alloc((size_t)F * 2);
    // f16 scratch: XH up to F*192 halves; WT up to 256*256 halves
    _Float16* XH = (_Float16*)alloc(((size_t)F * 192 + 1) / 2 + 16);
    _Float16* WT = (_Float16*)alloc((size_t)256 * 256 / 2);

    // 1) agent embedding: ax = relu(x[N,60] @ embW + embB) -> F0[N,64]
    launch_gemm(stream, x, embW, embB, F0, N, 60, 64, 1, XH, WT, 1);

    // 2) three GATv2 layers on agent graph
    run_gat(stream, F0,  64, 4, Wl1, Wr1, a1, b1, ei, E, N,
            A, Bb, G, GMAXU, GSUM, F1, 1, XH, WT);
    run_gat(stream, F1, 256, 4, Wl2, Wr2, a2, b2, ei, E, N,
            A, Bb, G, GMAXU, GSUM, F0, 1, XH, WT);
    run_gat(stream, F0, 256, 1, Wl3, Wr3, a3, b3, ei, E, N,
            A, Bb, G, GMAXU, GSUM, H3, 0, XH, WT);

    // 3) lane graph: mx = relu(lane_x @ mWe + mbe) -> A[L,64]; GATv2 -> mf = F1
    launch_gemm(stream, lane_x, mWe, mbe, A, L, 2, 64, 1, XH, WT, 1);
    run_gat(stream, A, 64, 1, Wlm, Wrm, am, bm, lei, EL, L,
            Bb, F0, G, GMAXU, GSUM, F1, 0, XH, WT);

    // 4) map global mean of mf[L,64] -> MEAN (scaled by 1/L at comb build)
    k_fill_u32<<<1, 64, 0, stream>>>((unsigned*)MEAN, 0u, 64);
    k_colsum<<<256, 64, 0, stream>>>(F1, L, 64, MEAN);

    // 5) centerline features: mean over P -> CLM[F,2]; CLM @ clW + clb -> F0[F,64]
    k_clmean<<<gsz((long)F * 2, 256), 256, 0, stream>>>(cl, F, P, CLM);
    launch_gemm(stream, CLM, clW, clb, F0, F, 2, 64, 0, XH, WT, 1);

    // 6) comb[F,192] = [h3[focal], map_mean, clf]  (spans A..Bb region)
    k_comb<<<gsz((long)F * 192, 256), 256, 0, stream>>>(H3, focal, MEAN,
                                                        1.0f / (float)L, F0, A, F);

    // 7) MLP head: z1 = relu(comb@f1W+f1b) -> F1; z2 = relu(z1@f2W+f2b) -> F0;
    //    out = z2@foW + fob
    launch_gemm(stream, A,  f1W, f1b, F1, F, 192, 64, 1, XH, WT, 1);
    launch_gemm(stream, F1, f2W, f2b, F0, F,  64, 32, 1, XH, WT, 1);
    launch_gemm(stream, F0, foW, fob, out, F, 32, 60, 0, XH, WT, 1);
}
